// Attention_89859305766952
// MI455X (gfx1250) — compile-verified, matrix operand order checked by reading, not run on hardware
//
#include <hip/hip_runtime.h>
#include <hip/hip_bf16.h>

// ---------------------------------------------------------------------------
// CDNA5 (gfx1250) attention:  QKV proj -> RoPE -> GQA causal flash attn -> out proj
// bf16 WMMA (V_WMMA_F32_16X16X32_BF16), f32 accumulate.
// Double-buffered LDS staging (batched loads -> batched stores), packed 8B LDS
// stores, log2-domain softmax, permlane16 reductions, causal liveness guards.
// ---------------------------------------------------------------------------

typedef __attribute__((ext_vector_type(16))) __bf16 v16bf;
typedef __attribute__((ext_vector_type(4)))  __bf16 v4bf;
typedef __attribute__((ext_vector_type(8)))  float  v8f;

#define SEQ     2048
#define DIM     4096
#define NH      32
#define NKV     8
#define HD      128
#define KVD     (NKV * HD)     // 1024
// 1/sqrt(128) * log2(e): softmax computed in base-2 domain.
#define SM_SCALE_LOG2 (0.0883883476483184f * 1.4426950408889634f)

#if __has_builtin(__builtin_amdgcn_exp2f)
#define EXP2F(x) __builtin_amdgcn_exp2f(x)
#else
#define EXP2F(x) exp2f(x)
#endif

// ---------------------------------------------------------------------------
// Fragment gather helpers.
// 16-bit A-matrix 16x32 layout (ISA 7.12.2): lane L holds row M = L%16;
// element e of the v16bf maps to K = e + (e>=8 ? 8 : 0) + 8*(L>=16).
// => 4 contiguous groups of 4: K in {0,4,16,20} + 8*laneHalf.
// Dense B-matrix (32x16) assumed to mirror this with N = L%16.
// ---------------------------------------------------------------------------
static __device__ __forceinline__ v16bf lds_frag(const __bf16* base, int row,
                                                 int ldm, int hf) {
  v16bf f;
  const __bf16* p = base + row * ldm + hf * 8;
  v4bf c0 = *reinterpret_cast<const v4bf*>(p + 0);
  v4bf c1 = *reinterpret_cast<const v4bf*>(p + 4);
  v4bf c2 = *reinterpret_cast<const v4bf*>(p + 16);
  v4bf c3 = *reinterpret_cast<const v4bf*>(p + 20);
  f[0]=c0[0];  f[1]=c0[1];  f[2]=c0[2];  f[3]=c0[3];
  f[4]=c1[0];  f[5]=c1[1];  f[6]=c1[2];  f[7]=c1[3];
  f[8]=c2[0];  f[9]=c2[1];  f[10]=c2[2]; f[11]=c2[3];
  f[12]=c3[0]; f[13]=c3[1]; f[14]=c3[2]; f[15]=c3[3];
  return f;
}

// Q fragment straight from global f32, with scale folded in.
static __device__ __forceinline__ v16bf global_frag_f32_scaled(const float* rowbase,
                                                               int hf, float scale) {
  v16bf f;
  const float* p = rowbase + hf * 8;
  float4 c0 = *reinterpret_cast<const float4*>(p + 0);
  float4 c1 = *reinterpret_cast<const float4*>(p + 4);
  float4 c2 = *reinterpret_cast<const float4*>(p + 16);
  float4 c3 = *reinterpret_cast<const float4*>(p + 20);
  f[0]=(__bf16)(c0.x*scale);  f[1]=(__bf16)(c0.y*scale);
  f[2]=(__bf16)(c0.z*scale);  f[3]=(__bf16)(c0.w*scale);
  f[4]=(__bf16)(c1.x*scale);  f[5]=(__bf16)(c1.y*scale);
  f[6]=(__bf16)(c1.z*scale);  f[7]=(__bf16)(c1.w*scale);
  f[8]=(__bf16)(c2.x*scale);  f[9]=(__bf16)(c2.y*scale);
  f[10]=(__bf16)(c2.z*scale); f[11]=(__bf16)(c2.w*scale);
  f[12]=(__bf16)(c3.x*scale); f[13]=(__bf16)(c3.y*scale);
  f[14]=(__bf16)(c3.z*scale); f[15]=(__bf16)(c3.w*scale);
  return f;
}

// ---------------------------------------------------------------------------
// 16-lane row reductions on the VALU pipe (v_permlane16_b32).
// Row-relative xor selects: nibble i of (sel1:sel0) = i ^ mask.
// ---------------------------------------------------------------------------
#if __has_builtin(__builtin_amdgcn_permlane16)
#define PERMXOR(v, S0, S1)                                                    \
  __int_as_float(__builtin_amdgcn_permlane16(__float_as_int(v),               \
                 __float_as_int(v), (int)(S0), (int)(S1), false, false))
static __device__ __forceinline__ float hmax16(float v) {
  v = fmaxf(v, PERMXOR(v, 0x67452301u, 0xefcdab89u));  // xor 1
  v = fmaxf(v, PERMXOR(v, 0x54761032u, 0xdcfe98bau));  // xor 2
  v = fmaxf(v, PERMXOR(v, 0x32107654u, 0xba98fedcu));  // xor 4
  v = fmaxf(v, PERMXOR(v, 0xfedcba98u, 0x76543210u));  // xor 8
  return v;
}
static __device__ __forceinline__ float hsum16(float v) {
  v += PERMXOR(v, 0x67452301u, 0xefcdab89u);
  v += PERMXOR(v, 0x54761032u, 0xdcfe98bau);
  v += PERMXOR(v, 0x32107654u, 0xba98fedcu);
  v += PERMXOR(v, 0xfedcba98u, 0x76543210u);
  return v;
}
#else
static __device__ __forceinline__ float hmax16(float v) {
  v = fmaxf(v, __shfl_xor(v, 1, 32));
  v = fmaxf(v, __shfl_xor(v, 2, 32));
  v = fmaxf(v, __shfl_xor(v, 4, 32));
  v = fmaxf(v, __shfl_xor(v, 8, 32));
  return v;
}
static __device__ __forceinline__ float hsum16(float v) {
  v += __shfl_xor(v, 1, 32);
  v += __shfl_xor(v, 2, 32);
  v += __shfl_xor(v, 4, 32);
  v += __shfl_xor(v, 8, 32);
  return v;
}
#endif

// ---------------------------------------------------------------------------
// GEMM: C[M,N] = A[M,K] * B[K,N], f32 in/out, bf16 WMMA compute.
// WG tile 128x128, 4 waves (2x2), each wave 64x64 (4x4 accumulators).
// Staging is two-phase: batch ALL global loads into registers (clause with
// descending loadcnt waits), then convert + packed ds_store_b64.
// ---------------------------------------------------------------------------
template <int N, int K>
static __device__ __forceinline__ void stage_tiles(const float* __restrict__ A,
                                                   const float* __restrict__ B,
                                                   __bf16* sA, __bf16* sB,
                                                   int m0, int n0, int kb, int tid) {
  float4 va[8];
  float4 vb[2][4];
  // ---- phase 1: issue all global loads ----
#pragma unroll
  for (int i = 0; i < 8; ++i) {
    int lin = tid + i * 128;                  // 0..1023 float4 groups
    int r = lin >> 3, cg = (lin & 7) * 4;
    va[i] = *reinterpret_cast<const float4*>(A + (size_t)(m0 + r) * K + kb + cg);
  }
#pragma unroll
  for (int b = 0; b < 2; ++b) {
    int bid = tid + b * 128;                  // 0..255 4x4 blocks
    int nb = bid & 31, k4 = bid >> 5;
    const float* bp = B + (size_t)(kb + k4 * 4) * N + n0 + nb * 4;
#pragma unroll
    for (int j = 0; j < 4; ++j)
      vb[b][j] = *reinterpret_cast<const float4*>(bp + j * N);
  }
  // ---- phase 2: convert + packed LDS stores ----
#pragma unroll
  for (int i = 0; i < 8; ++i) {
    int lin = tid + i * 128;
    int r = lin >> 3, cg = (lin & 7) * 4;
    v4bf pk; pk[0]=(__bf16)va[i].x; pk[1]=(__bf16)va[i].y;
    pk[2]=(__bf16)va[i].z; pk[3]=(__bf16)va[i].w;
    *reinterpret_cast<v4bf*>(&sA[r * 32 + cg]) = pk;
  }
#pragma unroll
  for (int b = 0; b < 2; ++b) {
    int bid = tid + b * 128;
    int nb = bid & 31, k4 = bid >> 5;
    v4bf p0; p0[0]=(__bf16)vb[b][0].x; p0[1]=(__bf16)vb[b][1].x; p0[2]=(__bf16)vb[b][2].x; p0[3]=(__bf16)vb[b][3].x;
    v4bf p1; p1[0]=(__bf16)vb[b][0].y; p1[1]=(__bf16)vb[b][1].y; p1[2]=(__bf16)vb[b][2].y; p1[3]=(__bf16)vb[b][3].y;
    v4bf p2; p2[0]=(__bf16)vb[b][0].z; p2[1]=(__bf16)vb[b][1].z; p2[2]=(__bf16)vb[b][2].z; p2[3]=(__bf16)vb[b][3].z;
    v4bf p3; p3[0]=(__bf16)vb[b][0].w; p3[1]=(__bf16)vb[b][1].w; p3[2]=(__bf16)vb[b][2].w; p3[3]=(__bf16)vb[b][3].w;
    *reinterpret_cast<v4bf*>(&sB[(nb * 4 + 0) * 32 + k4 * 4]) = p0;
    *reinterpret_cast<v4bf*>(&sB[(nb * 4 + 1) * 32 + k4 * 4]) = p1;
    *reinterpret_cast<v4bf*>(&sB[(nb * 4 + 2) * 32 + k4 * 4]) = p2;
    *reinterpret_cast<v4bf*>(&sB[(nb * 4 + 3) * 32 + k4 * 4]) = p3;
  }
}

template <int N, int K>
__global__ __launch_bounds__(128) void gemm_bf16(const float* __restrict__ A,
                                                 const float* __restrict__ B,
                                                 float* __restrict__ C) {
  __shared__ __align__(16) __bf16 sA[2][128 * 32];
  __shared__ __align__(16) __bf16 sB[2][128 * 32];

  const int tid  = threadIdx.x;
  const int wave = tid >> 5;
  const int lane = tid & 31;
  const int hf   = lane >> 4;
  const int l16  = lane & 15;
  const int m0 = blockIdx.y * 128;
  const int n0 = blockIdx.x * 128;
  const int wr = wave >> 1, wc = wave & 1;

  v8f acc[4][4];
#pragma unroll
  for (int i = 0; i < 4; ++i)
#pragma unroll
    for (int j = 0; j < 4; ++j) acc[i][j] = {};

  stage_tiles<N, K>(A, B, sA[0], sB[0], m0, n0, 0, tid);
  __syncthreads();

  int p = 0;
  for (int kb = 0; kb < K; kb += 32) {
    if (kb + 32 < K)
      stage_tiles<N, K>(A, B, sA[p ^ 1], sB[p ^ 1], m0, n0, kb + 32, tid);
    if (kb + 64 < K) {   // prime L2 two tiles ahead (global_prefetch_b8)
      __builtin_prefetch(A + (size_t)(m0 + (tid >> 3)) * K + kb + 64 + (tid & 7) * 4, 0, 3);
      __builtin_prefetch(B + (size_t)(kb + 64 + (tid >> 5)) * N + n0 + (tid & 31) * 4, 0, 3);
    }

    v16bf af[4], bf[4];
#pragma unroll
    for (int i = 0; i < 4; ++i) af[i] = lds_frag(sA[p], wr * 64 + i * 16 + l16, 32, hf);
#pragma unroll
    for (int j = 0; j < 4; ++j) bf[j] = lds_frag(sB[p], wc * 64 + j * 16 + l16, 32, hf);

#pragma unroll
    for (int i = 0; i < 4; ++i)
#pragma unroll
      for (int j = 0; j < 4; ++j)
        acc[i][j] = __builtin_amdgcn_wmma_f32_16x16x32_bf16(false, af[i], false, bf[j],
                                                            (short)0, acc[i][j], false, false);
    __syncthreads();
    p ^= 1;
  }

  float* cptr = C + (size_t)(m0 + wr * 64 + 8 * hf) * N + (n0 + wc * 64 + l16);
#pragma unroll
  for (int i = 0; i < 4; ++i)
#pragma unroll
    for (int j = 0; j < 4; ++j)
#pragma unroll
      for (int e = 0; e < 8; ++e)
        cptr[(size_t)(i * 16 + e) * N + j * 16] = acc[i][j][e];
}

// ---------------------------------------------------------------------------
// RoPE (interleaved pairs), in place on [seq, n_heads, 128] f32.
// ---------------------------------------------------------------------------
__global__ void rope_kernel(float* __restrict__ buf, int n_heads, int seqlen) {
  int idx = blockIdx.x * blockDim.x + threadIdx.x;
  int total = seqlen * n_heads * (HD / 2);
  if (idx >= total) return;
  int j   = idx & 63;
  int t   = idx >> 6;
  int hh  = t % n_heads;
  int pos = t / n_heads;
  float inv_freq = __expf(-(float)(2 * j) * (1.0f / 128.0f) * 9.210340371976184f);
  float ang = (float)pos * inv_freq;
  float s, c;
  __sincosf(ang, &s, &c);
  float* p = buf + ((size_t)pos * n_heads + hh) * HD + 2 * j;
  float xr = p[0], xi = p[1];
  p[0] = xr * c - xi * s;
  p[1] = xr * s + xi * c;
}

// ---------------------------------------------------------------------------
// Flash attention: grid (seq/128, NH), 128 threads (4 waves).
// Each wave owns 32 query rows (2 strips of 16); KV step 32, double-buffered.
// Two-phase staging; per-strip liveness guards skip fully-masked tiles.
// ---------------------------------------------------------------------------
static __device__ __forceinline__ void stage_kv(const float* __restrict__ Kb,
                                                const float* __restrict__ Vb,
                                                __bf16* sK, __bf16* sV,
                                                int kv0, int kvbase, int tid) {
  float4 kk[8];
  float4 vv[2][4];
  // ---- phase 1: issue all global loads ----
#pragma unroll
  for (int i = 0; i < 8; ++i) {
    int lin = tid + i * 128;
    int kv = lin >> 5;
    int d  = (lin & 31) * 4;
    kk[i] = *reinterpret_cast<const float4*>(Kb + (size_t)(kv0 + kv) * KVD + kvbase + d);
  }
#pragma unroll
  for (int b = 0; b < 2; ++b) {
    int bid = tid + b * 128;
    int db = bid & 31, kv4 = bid >> 5;
    const float* vp = Vb + (size_t)(kv0 + kv4 * 4) * KVD + kvbase + db * 4;
#pragma unroll
    for (int j = 0; j < 4; ++j)
      vv[b][j] = *reinterpret_cast<const float4*>(vp + j * KVD);
  }
  // ---- phase 2: convert + packed LDS stores ----
#pragma unroll
  for (int i = 0; i < 8; ++i) {
    int lin = tid + i * 128;
    int kv = lin >> 5;
    int d  = (lin & 31) * 4;
    v4bf pk; pk[0]=(__bf16)kk[i].x; pk[1]=(__bf16)kk[i].y;
    pk[2]=(__bf16)kk[i].z; pk[3]=(__bf16)kk[i].w;
    *reinterpret_cast<v4bf*>(&sK[kv * 136 + d]) = pk;
  }
#pragma unroll
  for (int b = 0; b < 2; ++b) {
    int bid = tid + b * 128;
    int db = bid & 31, kv4 = bid >> 5;
    v4bf p0; p0[0]=(__bf16)vv[b][0].x; p0[1]=(__bf16)vv[b][1].x; p0[2]=(__bf16)vv[b][2].x; p0[3]=(__bf16)vv[b][3].x;
    v4bf p1; p1[0]=(__bf16)vv[b][0].y; p1[1]=(__bf16)vv[b][1].y; p1[2]=(__bf16)vv[b][2].y; p1[3]=(__bf16)vv[b][3].y;
    v4bf p2; p2[0]=(__bf16)vv[b][0].z; p2[1]=(__bf16)vv[b][1].z; p2[2]=(__bf16)vv[b][2].z; p2[3]=(__bf16)vv[b][3].z;
    v4bf p3; p3[0]=(__bf16)vv[b][0].w; p3[1]=(__bf16)vv[b][1].w; p3[2]=(__bf16)vv[b][2].w; p3[3]=(__bf16)vv[b][3].w;
    *reinterpret_cast<v4bf*>(&sV[(db * 4 + 0) * 40 + kv4 * 4]) = p0;
    *reinterpret_cast<v4bf*>(&sV[(db * 4 + 1) * 40 + kv4 * 4]) = p1;
    *reinterpret_cast<v4bf*>(&sV[(db * 4 + 2) * 40 + kv4 * 4]) = p2;
    *reinterpret_cast<v4bf*>(&sV[(db * 4 + 3) * 40 + kv4 * 4]) = p3;
  }
}

__global__ __launch_bounds__(128) void flash_attn(const float* __restrict__ Q,
                                                  const float* __restrict__ Kb,
                                                  const float* __restrict__ Vb,
                                                  float* __restrict__ O) {
  __shared__ __align__(16) __bf16 sK[2][32 * 136];     // [kv][d], padded
  __shared__ __align__(16) __bf16 sV[2][128 * 40];     // transposed [d][kv], padded
  __shared__ __align__(16) __bf16 sP[8 * 16 * 32];     // per-(wave,strip) P patch

  const int tid  = threadIdx.x;
  const int wave = tid >> 5;
  const int lane = tid & 31;
  const int hf   = lane >> 4;
  const int l16  = lane & 15;
  const int h    = blockIdx.y;
  const int kvh  = h >> 2;                 // N_REP = 4
  const int kvbase = kvh * HD;
  const int q0   = blockIdx.x * 128;
  const int qw   = q0 + wave * 32;

  // Q fragments (scale * log2e folded in): 2 strips x 4 chunks of K=32.
  v16bf qf[2][4];
#pragma unroll
  for (int st = 0; st < 2; ++st) {
    const float* qrow = Q + (size_t)(qw + st * 16 + l16) * DIM + h * HD;
#pragma unroll
    for (int ch = 0; ch < 4; ++ch)
      qf[st][ch] = global_frag_f32_scaled(qrow + ch * 32, hf, SM_SCALE_LOG2);
  }

  v8f o[2][8];
  float mrow[2][8], lrow[2][8];
#pragma unroll
  for (int st = 0; st < 2; ++st)
#pragma unroll
    for (int e = 0; e < 8; ++e) {
      o[st][e] = {};
      mrow[st][e] = -3.0e38f;
      lrow[st][e] = 0.0f;
    }

  const int kv_end = q0 + 128;
  stage_kv(Kb, Vb, sK[0], sV[0], 0, kvbase, tid);
  __syncthreads();

  int p = 0;
  for (int kv0 = 0; kv0 < kv_end; kv0 += 32) {
    if (kv0 + 32 < kv_end)
      stage_kv(Kb, Vb, sK[p ^ 1], sV[p ^ 1], kv0 + 32, kvbase, tid);

    // wave-uniform liveness: strip st (rows qw+st*16 .. +15) has unmasked
    // elements iff kv0 <= last row of strip.
    const bool live0 = kv0 <= qw + 15;
    const bool live1 = kv0 <= qw + 31;

    if (live1) {
      // ---- S = (scaled Q) K^T: K fragments shared by live strips ----
      v8f s[2][2];
      s[0][0] = {}; s[0][1] = {}; s[1][0] = {}; s[1][1] = {};
#pragma unroll
      for (int sub = 0; sub < 2; ++sub)
#pragma unroll
        for (int ch = 0; ch < 4; ++ch) {
          v16bf kf = lds_frag(sK[p] + ch * 32, sub * 16 + l16, 136, hf);
          if (live0)
            s[0][sub] = __builtin_amdgcn_wmma_f32_16x16x32_bf16(false, qf[0][ch], false, kf,
                                                                (short)0, s[0][sub], false, false);
          s[1][sub] = __builtin_amdgcn_wmma_f32_16x16x32_bf16(false, qf[1][ch], false, kf,
                                                              (short)0, s[1][sub], false, false);
        }

      // ---- per-strip: causal mask (diagonal only) + online softmax ----
      v16bf pf[2];
#pragma unroll
      for (int st = 0; st < 2; ++st) {
        if (st == 0 && !live0) continue;
        if (kv0 + 31 > qw + st * 16) {       // block touches the diagonal
#pragma unroll
          for (int e = 0; e < 8; ++e) {
            int q = qw + st * 16 + e + 8 * hf;
#pragma unroll
            for (int sub = 0; sub < 2; ++sub) {
              int kv = kv0 + sub * 16 + l16;
              if (kv > q) s[st][sub][e] = -3.0e38f;
            }
          }
        }
        float tmax[8];
#pragma unroll
        for (int e = 0; e < 8; ++e)
          tmax[e] = hmax16(fmaxf(s[st][0][e], s[st][1][e]));
#pragma unroll
        for (int e = 0; e < 8; ++e) {
          float mn  = fmaxf(mrow[st][e], tmax[e]);
          float scl = EXP2F(mrow[st][e] - mn);
          float p0  = EXP2F(s[st][0][e] - mn);
          float p1  = EXP2F(s[st][1][e] - mn);
          s[st][0][e] = p0; s[st][1][e] = p1;
          float rs = hsum16(p0 + p1);
          lrow[st][e] = lrow[st][e] * scl + rs;
          mrow[st][e] = mn;
#pragma unroll
          for (int t = 0; t < 8; ++t) o[st][t][e] *= scl;
        }
        __bf16* pbuf = sP + (wave * 2 + st) * (16 * 32);
#pragma unroll
        for (int sub = 0; sub < 2; ++sub)
#pragma unroll
          for (int e = 0; e < 8; ++e)
            pbuf[(e + 8 * hf) * 32 + sub * 16 + l16] = (__bf16)s[st][sub][e];
        asm volatile("" ::: "memory");  // same-wave DS ops are HW-ordered
        pf[st] = lds_frag(pbuf, l16, 32, hf);
      }

      // ---- O += P * V: V fragments shared by live strips ----
#pragma unroll
      for (int t = 0; t < 8; ++t) {
        v16bf vf = lds_frag(sV[p], t * 16 + l16, 40, hf);
        if (live0)
          o[0][t] = __builtin_amdgcn_wmma_f32_16x16x32_bf16(false, pf[0], false, vf,
                                                            (short)0, o[0][t], false, false);
        o[1][t] = __builtin_amdgcn_wmma_f32_16x16x32_bf16(false, pf[1], false, vf,
                                                          (short)0, o[1][t], false, false);
      }
    }
    __syncthreads();
    p ^= 1;
  }

  // ---- normalize and write [seq, NH*HD] ----
#pragma unroll
  for (int st = 0; st < 2; ++st)
#pragma unroll
    for (int e = 0; e < 8; ++e) {
      float inv = 1.0f / lrow[st][e];
      int row = qw + st * 16 + e + 8 * hf;
#pragma unroll
      for (int t = 0; t < 8; ++t)
        O[(size_t)row * DIM + h * HD + t * 16 + l16] = o[st][t][e] * inv;
    }
}

// ---------------------------------------------------------------------------
// Launch
// ---------------------------------------------------------------------------
extern "C" void kernel_launch(void* const* d_in, const int* in_sizes, int n_in,
                              void* d_out, int out_size, void* d_ws, size_t ws_size,
                              hipStream_t stream) {
  const float* x  = (const float*)d_in[0];
  const float* wq = (const float*)d_in[1];
  const float* wk = (const float*)d_in[2];
  const float* wv = (const float*)d_in[3];
  const float* wo = (const float*)d_in[4];
  float* out = (float*)d_out;

  float* Q  = (float*)d_ws;                  // [2048, 4096]
  float* Kb = Q  + (size_t)SEQ * DIM;        // [2048, 1024]
  float* Vb = Kb + (size_t)SEQ * KVD;        // [2048, 1024]
  float* At = Vb + (size_t)SEQ * KVD;        // [2048, 4096]

  dim3 blk(128);
  gemm_bf16<DIM, DIM><<<dim3(DIM / 128, SEQ / 128), blk, 0, stream>>>(x, wq, Q);
  gemm_bf16<KVD, DIM><<<dim3(KVD / 128, SEQ / 128), blk, 0, stream>>>(x, wk, Kb);
  gemm_bf16<KVD, DIM><<<dim3(KVD / 128, SEQ / 128), blk, 0, stream>>>(x, wv, Vb);

  int nq = SEQ * NH  * (HD / 2);
  int nk = SEQ * NKV * (HD / 2);
  rope_kernel<<<(nq + 255) / 256, 256, 0, stream>>>(Q,  NH,  SEQ);
  rope_kernel<<<(nk + 255) / 256, 256, 0, stream>>>(Kb, NKV, SEQ);

  flash_attn<<<dim3(SEQ / 128, NH), blk, 0, stream>>>(Q, Kb, Vb, At);

  gemm_bf16<DIM, DIM><<<dim3(DIM / 128, SEQ / 128), blk, 0, stream>>>(At, wo, out);
}
